// MultiheadLatentAttentionTensorCast_55508157333986
// MI455X (gfx1250) — matrix operand order repaired and checked by reading, not compiled, not run on hardware
//
#include <hip/hip_runtime.h>
#include <hip/hip_bf16.h>

// ---------------------------------------------------------------------------
// MLA (multi-head latent attention) with fp32->bf16 tensor-cast, CDNA5 WMMA.
// Shapes: H=16, DN=128, DR=64, DV=128, L=512, B=2, S=2048, HID=2048, T=4096.
// ---------------------------------------------------------------------------

typedef __bf16 bf16_t;
typedef __attribute__((ext_vector_type(16))) bf16_t v16bf;
typedef __attribute__((ext_vector_type(8)))  bf16_t v8bf;
typedef __attribute__((ext_vector_type(8)))  float  v8f;

constexpr int H   = 16;
constexpr int DN  = 128;
constexpr int DR  = 64;
constexpr int DV  = 128;
constexpr int L   = 512;
constexpr int NB  = 2;          // batch
constexpr int S   = 2048;       // sequence
constexpr int HID = 2048;
constexpr int T   = NB * S;     // 4096 tokens
constexpr int DQ  = DN + DR;    // 192 per-head q dim
constexpr int DC  = L + DR;     // 576 latent+rope kv dim
constexpr float SCALE = 0.07216878364870322f;  // 1/sqrt(DN+DR)

// ---------------------------------------------------------------------------
// Elementwise fp32 -> bf16 cast
// ---------------------------------------------------------------------------
__global__ void cast_f32_bf16(const float* __restrict__ in, bf16_t* __restrict__ out, size_t n) {
  size_t i = (size_t)blockIdx.x * blockDim.x + threadIdx.x;
  if (i < n) out[i] = (bf16_t)in[i];
}

// Batched 2D transpose + cast: in[z][r][c] (fp32) -> out[z][c][r] (bf16)
__global__ void transpose_cast_f32_bf16(const float* __restrict__ in, bf16_t* __restrict__ out,
                                        int R, int C) {
  int z = blockIdx.y;
  size_t n = (size_t)R * C;
  size_t idx = (size_t)blockIdx.x * blockDim.x + threadIdx.x;
  if (idx >= n) return;
  int r = (int)(idx / C), c = (int)(idx % C);
  out[(size_t)z * n + (size_t)c * R + r] = (bf16_t)in[(size_t)z * n + idx];
}

// kv_cat[t][c] (bf16, c<L) -> kv_t[c][t] (bf16) : V matrix pre-transpose
__global__ void transpose_kv(const bf16_t* __restrict__ kv_cat, bf16_t* __restrict__ kv_t) {
  size_t idx = (size_t)blockIdx.x * blockDim.x + threadIdx.x;
  if (idx >= (size_t)L * T) return;
  int t = (int)(idx / L), c = (int)(idx % L);
  kv_t[(size_t)c * T + t] = kv_cat[(size_t)t * DC + c];
}

// ---------------------------------------------------------------------------
// Generic WMMA bf16 GEMM (NT): C[m,n] = sum_k A[m,k] * B[n,k]
//   A row-major [M,K] bf16, B row-major [N,K] bf16, C row-major, fp32 acc.
//   Block = 256 threads = 8 waves -> 32x64 block tile (2x4 waves of 16x16).
//   grid = (N/64, M/32, batch). All dims divide evenly by construction.
//   Fragment layout per CDNA5 ISA 7.12.2:
//     A 16x32 bf16: lane l: m=l&15, h=l>>4; elem e: k = e+8h (e<8) / e+8+8h
//     B 32x16 bf16: lane l: n=l&15, h=l>>4; elem e: k = e+8h (e<8) / e+8+8h
//     C 16x16 f32 : lane l: n=l&15, h=l>>4; reg r: m = r+8h
// ---------------------------------------------------------------------------
template <typename CT>
__global__ __launch_bounds__(256) void gemm_nt_bf16(
    const bf16_t* __restrict__ A, const bf16_t* __restrict__ B, CT* __restrict__ C,
    int K, long lda, long ldb, long ldc, long sA, long sB, long sC) {
  int z  = blockIdx.z;
  const bf16_t* Ab = A + (size_t)z * sA;
  const bf16_t* Bb = B + (size_t)z * sB;
  CT*           Cb = C + (size_t)z * sC;

  int tid = threadIdx.x;
  int w = tid >> 5, l = tid & 31, lh = l & 15, hh = l >> 4;
  int m0 = blockIdx.y * 32 + (w & 1) * 16;
  int n0 = blockIdx.x * 64 + (w >> 1) * 16;

  const bf16_t* ar = Ab + (size_t)(m0 + lh) * lda;
  const bf16_t* br = Bb + (size_t)(n0 + lh) * ldb;

  v8f acc;
#pragma unroll
  for (int i = 0; i < 8; ++i) acc[i] = 0.0f;

  for (int k0 = 0; k0 < K; k0 += 32) {
    v8bf alo = *(const v8bf*)(ar + k0 + 8 * hh);
    v8bf ahi = *(const v8bf*)(ar + k0 + 16 + 8 * hh);
    v8bf blo = *(const v8bf*)(br + k0 + 8 * hh);
    v8bf bhi = *(const v8bf*)(br + k0 + 16 + 8 * hh);
    if (k0 + 32 < K) {
      __builtin_prefetch(ar + k0 + 64, 0, 1);
      __builtin_prefetch(br + k0 + 64, 0, 1);
    }
    v16bf a, b;
#pragma unroll
    for (int i = 0; i < 8; ++i) {
      a[i] = alo[i];  a[i + 8] = ahi[i];
      b[i] = blo[i];  b[i + 8] = bhi[i];
    }
    acc = __builtin_amdgcn_wmma_f32_16x16x32_bf16(false, a, false, b, (short)0, acc,
                                                  false, false);
  }
#pragma unroll
  for (int r = 0; r < 8; ++r)
    Cb[(size_t)(m0 + r + 8 * hh) * ldc + n0 + lh] = (CT)acc[r];
}

// ---------------------------------------------------------------------------
// Fused RMSNorm (over L=512) + k-RoPE (DR=64, with (32,2)-interleave perm).
// One block (256 threads) per token. ckv is fp32 [T, 576].
// ---------------------------------------------------------------------------
__global__ __launch_bounds__(256) void rmsnorm_krope(
    const float* __restrict__ ckv, const float* __restrict__ ln_w,
    const float* __restrict__ cosb, const float* __restrict__ sinb,
    bf16_t* __restrict__ kv_cat) {
  int t = blockIdx.x, i = threadIdx.x;
  const float* row = ckv + (size_t)t * DC;
  float v1 = row[i], v2 = row[i + 256];

  __shared__ float red[256];
  red[i] = v1 * v1 + v2 * v2;
  __syncthreads();
  for (int s2 = 128; s2 > 0; s2 >>= 1) {
    if (i < s2) red[i] += red[i + s2];
    __syncthreads();
  }
  float inv = rsqrtf(red[0] / (float)L + 1e-6f);

  kv_cat[(size_t)t * DC + i]       = (bf16_t)(v1 * inv * ln_w[i]);
  kv_cat[(size_t)t * DC + i + 256] = (bf16_t)(v2 * inv * ln_w[i + 256]);

  if (i < DR) {
    int j = i;
    float c = cosb[(size_t)t * DR + j], sn = sinb[(size_t)t * DR + j];
    float o;
    if (j < 32) {
      float a = row[L + 2 * j], b = row[L + 2 * j + 1];
      o = a * c - b * sn;                 // p[j]=r[2j], rotate_half -> -p[j+32]=-r[2j+1]
    } else {
      int jj = j - 32;
      float a = row[L + 2 * jj], b = row[L + 2 * jj + 1];
      o = b * c + a * sn;                 // p[j]=r[2jj+1], rotate_half -> p[j-32]=r[2jj]
    }
    kv_cat[(size_t)t * DC + L + j] = (bf16_t)o;
  }
}

// q-RoPE: q[T,H,192] bf16 -> q_cat[T,H,576] rope section (cols 512..575)
__global__ void qrope(const bf16_t* __restrict__ q, const float* __restrict__ cosb,
                      const float* __restrict__ sinb, bf16_t* __restrict__ q_cat) {
  size_t idx = (size_t)blockIdx.x * blockDim.x + threadIdx.x;  // T*H*DR total
  int j = (int)(idx % DR);
  int h = (int)((idx / DR) % H);
  int t = (int)(idx / ((size_t)DR * H));
  const bf16_t* r = q + ((size_t)t * H + h) * DQ + DN;
  float c = cosb[(size_t)t * DR + j], sn = sinb[(size_t)t * DR + j];
  float o;
  if (j < 32) {
    float a = (float)r[2 * j], b = (float)r[2 * j + 1];
    o = a * c - b * sn;
  } else {
    int jj = j - 32;
    float a = (float)r[2 * jj], b = (float)r[2 * jj + 1];
    o = b * c + a * sn;
  }
  q_cat[((size_t)t * H + h) * DC + L + j] = (bf16_t)o;
}

// ---------------------------------------------------------------------------
// Flash attention over latent space (cooperative, no redundant WMMA).
//   scores[q,k] = q_cat[q,h,0:576] . kv_cat[k,0:576] * scale  (causal)
//   out_lat[q,h,0:512] = softmax(scores) @ kv_cat[:,0:512]
//
// Block = 256 thr = 8 waves; 16 queries per block; key tiles of 32 keys.
// Per key tile:
//   1. Score K-dim (576 = 18 fragments) is SPLIT across waves: wave w takes
//      fragments {w, w+8} plus (for waves 0-1) a tail fragment {16+w}.
//      The wave index is readfirstlane'd so the tail test is a *scalar*
//      branch: no EXEC masking ever wraps a WMMA (ISA requires EXEC all-1s).
//   2. Partials land in per-wave LDS slices (f32 16x32); after a barrier,
//      wave w owns the online softmax of query rows {2w, 2w+1}: sums the 8
//      partials, applies scale+causal mask, full-wave shfl_xor row max/sum,
//      keeps running (m,l), publishes exp-P (bf16 row-major) + corr to LDS.
//   3. After a second barrier every wave rescales its private 16x64 f32
//      accumulator by corr[row], reads P as an A fragment (2x ds b128) and
//      its V columns from pre-transposed kv_t (contiguous b128 B fragments),
//      then issues its 4 unique attn-x-V WMMAs.
// ---------------------------------------------------------------------------
__global__ __launch_bounds__(256) void mla_attn(
    const bf16_t* __restrict__ q_cat, const bf16_t* __restrict__ kv_cat,
    const bf16_t* __restrict__ kv_t, bf16_t* __restrict__ out_lat) {
  __shared__ float  sbuf[8][16][32];                                // 16 KB partial scores
  __shared__ bf16_t pbuf[16][32] __attribute__((aligned(16)));      // 1 KB exp-P
  __shared__ float  corrbuf[16];
  __shared__ float  lbuf[16];

  int tid = threadIdx.x;
  int w  = __builtin_amdgcn_readfirstlane(tid >> 5);   // scalar wave id
  int l  = tid & 31, lh = l & 15, hh = l >> 4;
  int q0 = blockIdx.x * 16, head = blockIdx.y, bz = blockIdx.z;

  const bf16_t* qptr = q_cat + ((size_t)(bz * S + q0 + lh) * H + head) * DC;
  const bf16_t* kbp  = kv_cat + (size_t)(bz * S) * DC;
  const bf16_t* vbp  = kv_t + (size_t)(bz * S);

  // softmax-owner state: wave w owns query rows {2w, 2w+1} (scalar-derived)
  const int row0 = 2 * w, row1 = 2 * w + 1;
  float m0 = -1e30f, m1 = -1e30f, l0 = 0.0f, l1 = 0.0f;

  v8f acc[4];
#pragma unroll
  for (int vt = 0; vt < 4; ++vt)
#pragma unroll
    for (int r = 0; r < 8; ++r) acc[vt][r] = 0.0f;

  int nkt = (q0 + 15) / 32 + 1;
  for (int kt = 0; kt < nkt; ++kt) {
    int kb = kt * 32;

    // --- 1. partial scores: this wave's K-fragment subset ---
    v8f s0, s1;
#pragma unroll
    for (int i = 0; i < 8; ++i) { s0[i] = 0.0f; s1[i] = 0.0f; }

    const bf16_t* k0p = kbp + (size_t)(kb + lh) * DC + 8 * hh;
    const bf16_t* k1p = kbp + (size_t)(kb + 16 + lh) * DC + 8 * hh;

    auto score_frag = [&](int ks8) {
      int ks = ks8 * 32;
      v8bf alo = *(const v8bf*)(qptr + ks + 8 * hh);
      v8bf ahi = *(const v8bf*)(qptr + ks + 16 + 8 * hh);
      v8bf b0l = *(const v8bf*)(k0p + ks);
      v8bf b0h = *(const v8bf*)(k0p + ks + 16);
      v8bf b1l = *(const v8bf*)(k1p + ks);
      v8bf b1h = *(const v8bf*)(k1p + ks + 16);
      v16bf a, b0, b1;
#pragma unroll
      for (int i = 0; i < 8; ++i) {
        a[i] = alo[i];  a[i + 8] = ahi[i];
        b0[i] = b0l[i]; b0[i + 8] = b0h[i];
        b1[i] = b1l[i]; b1[i + 8] = b1h[i];
      }
      s0 = __builtin_amdgcn_wmma_f32_16x16x32_bf16(false, a, false, b0, (short)0, s0,
                                                   false, false);
      s1 = __builtin_amdgcn_wmma_f32_16x16x32_bf16(false, a, false, b1, (short)0, s1,
                                                   false, false);
    };
    score_frag(w);          // fragments 0..7
    score_frag(w + 8);      // fragments 8..15
    if (w < 2)              // scalar-uniform branch: tail fragments 16,17
      score_frag(16 + w);

    // store partials (C layout: row = r+8*hh, key col = lane&15 / 16+lane&15)
#pragma unroll
    for (int r = 0; r < 8; ++r) {
      sbuf[w][r + 8 * hh][lh]      = s0[r];
      sbuf[w][r + 8 * hh][16 + lh] = s1[r];
    }
    __syncthreads();

    // --- 2. owner reduce + online softmax for rows {row0,row1}, col = lane ---
    {
      float x0 = 0.0f, x1 = 0.0f;
#pragma unroll
      for (int u = 0; u < 8; ++u) {
        x0 += sbuf[u][row0][l];
        x1 += sbuf[u][row1][l];
      }
      int key = kb + l;
      bool va = (key <= q0 + row0), vb = (key <= q0 + row1);
      float t0 = va ? x0 * SCALE : -1e30f;
      float t1 = vb ? x1 * SCALE : -1e30f;
      float mx0 = t0, mx1 = t1;
#pragma unroll
      for (int off = 1; off < 32; off <<= 1) {
        mx0 = fmaxf(mx0, __shfl_xor(mx0, off));
        mx1 = fmaxf(mx1, __shfl_xor(mx1, off));
      }
      float mn0 = fmaxf(m0, mx0), mn1 = fmaxf(m1, mx1);
      float c0 = __expf(m0 - mn0), c1 = __expf(m1 - mn1);
      float p0 = va ? __expf(t0 - mn0) : 0.0f;
      float p1 = vb ? __expf(t1 - mn1) : 0.0f;
      float ss0 = p0, ss1 = p1;
#pragma unroll
      for (int off = 1; off < 32; off <<= 1) {
        ss0 += __shfl_xor(ss0, off);
        ss1 += __shfl_xor(ss1, off);
      }
      l0 = l0 * c0 + ss0;
      l1 = l1 * c1 + ss1;
      m0 = mn0;
      m1 = mn1;
      pbuf[row0][l] = (bf16_t)p0;
      pbuf[row1][l] = (bf16_t)p1;
      if (l == 0) { corrbuf[row0] = c0; corrbuf[row1] = c1; }
    }
    __syncthreads();

    // --- 3. consume: rescale acc, P @ V for this wave's 64 V columns ---
    float crd[8];
#pragma unroll
    for (int r = 0; r < 8; ++r) crd[r] = corrbuf[r + 8 * hh];
#pragma unroll
    for (int vt = 0; vt < 4; ++vt)
#pragma unroll
      for (int r = 0; r < 8; ++r) acc[vt][r] *= crd[r];

    v8bf plo = *(const v8bf*)(&pbuf[lh][8 * hh]);
    v8bf phi = *(const v8bf*)(&pbuf[lh][16 + 8 * hh]);
    v16bf pf;
#pragma unroll
    for (int i = 0; i < 8; ++i) { pf[i] = plo[i]; pf[i + 8] = phi[i]; }

#pragma unroll
    for (int vt = 0; vt < 4; ++vt) {
      int col = w * 64 + vt * 16 + lh;
      const bf16_t* vp = vbp + (size_t)col * T + kb + 8 * hh;
      v8bf vlo = *(const v8bf*)(vp);
      v8bf vhi = *(const v8bf*)(vp + 16);
      v16bf vf;
#pragma unroll
      for (int i = 0; i < 8; ++i) { vf[i] = vlo[i]; vf[i + 8] = vhi[i]; }
      acc[vt] = __builtin_amdgcn_wmma_f32_16x16x32_bf16(false, pf, false, vf, (short)0,
                                                        acc[vt], false, false);
    }
  }

  // publish final softmax denominators, then normalize + store out_lat (bf16)
  if (l == 0) { lbuf[row0] = l0; lbuf[row1] = l1; }
  __syncthreads();
  float lf[8];
#pragma unroll
  for (int r = 0; r < 8; ++r) lf[r] = lbuf[r + 8 * hh];
#pragma unroll
  for (int vt = 0; vt < 4; ++vt)
#pragma unroll
    for (int r = 0; r < 8; ++r) {
      size_t t = (size_t)(bz * S + q0 + r + 8 * hh);
      out_lat[(t * H + head) * L + w * 64 + vt * 16 + lh] = (bf16_t)(acc[vt][r] / lf[r]);
    }
}

// ---------------------------------------------------------------------------
// Host-side launch
// ---------------------------------------------------------------------------
extern "C" void kernel_launch(void* const* d_in, const int* in_sizes, int n_in,
                              void* d_out, int out_size, void* d_ws, size_t ws_size,
                              hipStream_t stream) {
  (void)in_sizes; (void)n_in; (void)out_size; (void)ws_size;

  const float* x     = (const float*)d_in[0];   // [T, HID]
  const float* cosb  = (const float*)d_in[1];   // [T, DR]
  const float* sinb  = (const float*)d_in[2];   // [T, DR]
  const float* qw_f  = (const float*)d_in[3];   // [H*DQ, HID] = [3072, 2048]
  const float* kvw_f = (const float*)d_in[4];   // [DC, HID]   = [576, 2048]
  const float* ln_w  = (const float*)d_in[5];   // [L]
  const float* wukt  = (const float*)d_in[6];   // [H, DN, L]
  const float* wuv   = (const float*)d_in[7];   // [H, L, DV]
  const float* ow_f  = (const float*)d_in[8];   // [HID, H*DV] = [2048, 2048]
  float* out = (float*)d_out;                   // [T, HID]

  // --- workspace carve-out (all sizes multiples of 256 B) ---
  char* ws = (char*)d_ws;
  size_t off = 0;
  auto take = [&](size_t bytes) { char* p = ws + off; off += bytes; return p; };

  bf16_t* xb      = (bf16_t*)take((size_t)T * HID * 2);          // 16 MB
  bf16_t* qw      = (bf16_t*)take((size_t)H * DQ * HID * 2);     // 12 MB
  bf16_t* kvw     = (bf16_t*)take((size_t)DC * HID * 2);         // 2.25 MB
  bf16_t* ow      = (bf16_t*)take((size_t)HID * H * DV * 2);     // 8 MB
  bf16_t* wukt_t  = (bf16_t*)take((size_t)H * L * DN * 2);       // 2 MB  [H,L,DN]
  bf16_t* wuv_t   = (bf16_t*)take((size_t)H * DV * L * 2);       // 2 MB  [H,DV,L]
  bf16_t* qbuf    = (bf16_t*)take((size_t)T * H * DQ * 2);       // 24 MB [T,H*192]
  float*  ckv     = (float*) take((size_t)T * DC * 4);           // 9 MB  [T,576]
  bf16_t* kv_cat  = (bf16_t*)take((size_t)T * DC * 2);           // 4.5 MB
  bf16_t* q_cat   = (bf16_t*)take((size_t)T * H * DC * 2);       // 72 MB [T,H,576]
  bf16_t* kv_t    = (bf16_t*)take((size_t)L * T * 2);            // 4 MB  [L,T]
  bf16_t* out_lat = (bf16_t*)take((size_t)T * H * L * 2);        // 64 MB [T,H,512]
  bf16_t* att_out = (bf16_t*)take((size_t)T * H * DV * 2);       // 16 MB [T,2048]

  // --- 1. casts / weight transposes ---
  cast_f32_bf16<<<(T * (size_t)HID) / 256, 256, 0, stream>>>(x, xb, (size_t)T * HID);
  cast_f32_bf16<<<((size_t)H * DQ * HID) / 256, 256, 0, stream>>>(qw_f, qw, (size_t)H * DQ * HID);
  cast_f32_bf16<<<((size_t)DC * HID) / 256, 256, 0, stream>>>(kvw_f, kvw, (size_t)DC * HID);
  cast_f32_bf16<<<((size_t)HID * H * DV) / 256, 256, 0, stream>>>(ow_f, ow, (size_t)HID * H * DV);
  transpose_cast_f32_bf16<<<dim3((DN * L) / 256, H), 256, 0, stream>>>(wukt, wukt_t, DN, L);
  transpose_cast_f32_bf16<<<dim3((L * DV) / 256, H), 256, 0, stream>>>(wuv, wuv_t, L, DV);

  // --- 2. input projections (WMMA GEMMs, NT) ---
  // q = xb [T,HID] @ qw^T [H*DQ,HID] -> bf16 [T, H*DQ]
  gemm_nt_bf16<bf16_t><<<dim3((H * DQ) / 64, T / 32, 1), 256, 0, stream>>>(
      xb, qw, qbuf, HID, HID, HID, H * DQ, 0, 0, 0);
  // ckv = xb @ kvw^T -> fp32 [T, DC]
  gemm_nt_bf16<float><<<dim3(DC / 64, T / 32, 1), 256, 0, stream>>>(
      xb, kvw, ckv, HID, HID, HID, DC, 0, 0, 0);

  // --- 3. RMSNorm + RoPE ---
  rmsnorm_krope<<<T, 256, 0, stream>>>(ckv, ln_w, cosb, sinb, kv_cat);
  qrope<<<((size_t)T * H * DR) / 256, 256, 0, stream>>>(qbuf, cosb, sinb, q_cat);

  // --- 4. q_lat[t,h,l] = sum_d q_pass[t,h,d] * W_UK_T[h,d,l]  (per-head NT) ---
  // A = qbuf + h*DQ (lda=H*DQ), B = wukt_t[h] [L,DN] (ldb=DN), C = q_cat + h*DC
  gemm_nt_bf16<bf16_t><<<dim3(L / 64, T / 32, H), 256, 0, stream>>>(
      qbuf, wukt_t, q_cat, DN, H * DQ, DN, H * DC, DQ, (long)L * DN, DC);

  // --- 5. V transpose for contiguous B fragments ---
  transpose_kv<<<((size_t)L * T) / 256, 256, 0, stream>>>(kv_cat, kv_t);

  // --- 6. flash attention over latent space ---
  mla_attn<<<dim3(S / 16, H, NB), 256, 0, stream>>>(q_cat, kv_cat, kv_t, out_lat);

  // --- 7. value up-projection: out_v[t,h,v] = sum_l out_lat[t,h,l]*W_UV[h,l,v] ---
  // A = out_lat + h*L (lda=H*L), B = wuv_t[h] [DV,L] (ldb=L), C = att_out + h*DV
  gemm_nt_bf16<bf16_t><<<dim3(DV / 64, T / 32, H), 256, 0, stream>>>(
      out_lat, wuv_t, att_out, L, H * L, L, H * DV, L, (long)DV * L, DV);

  // --- 8. output projection: out = att_out [T,2048] @ ow^T [2048,2048] -> fp32 ---
  gemm_nt_bf16<float><<<dim3(HID / 64, T / 32, 1), 256, 0, stream>>>(
      att_out, ow, out, H * DV, H * DV, H * DV, HID, 0, 0, 0);
}
